// MinkowskiSpMiddleResNetFHD_7086696038820
// MI455X (gfx1250) — matrix operand order repaired and checked by reading, not compiled, not run on hardware
//
#include <hip/hip_runtime.h>
#include <stdint.h>

typedef unsigned short u16;
typedef __attribute__((ext_vector_type(16))) __bf16 v16bf;
typedef __attribute__((ext_vector_type(8)))  float  v8f;

union Frag16 { v16bf v; uint4 q[2]; u16 u[16]; };

// kernel tap offsets (dz,dy,dx), reference OFF27 / OFFZ order
__constant__ int c_off27[81] = {
  -1,-1,-1, -1,-1,0, -1,-1,1, -1,0,-1, -1,0,0, -1,0,1, -1,1,-1, -1,1,0, -1,1,1,
   0,-1,-1,  0,-1,0,  0,-1,1,  0,0,-1,  0,0,0,  0,0,1,  0,1,-1,  0,1,0,  0,1,1,
   1,-1,-1,  1,-1,0,  1,-1,1,  1,0,-1,  1,0,0,  1,0,1,  1,1,-1,  1,1,0,  1,1,1 };
__constant__ int c_offz[9] = { -1,0,0, 0,0,0, 1,0,0 };

__device__ __forceinline__ u16 f2bf(float f) {
  unsigned u = __float_as_uint(f);
  unsigned r = u + 0x7FFFu + ((u >> 16) & 1u);
  return (u16)(r >> 16);
}

// ---------------- small utility kernels ----------------

__global__ void k_f2bf(const float* __restrict__ x, u16* __restrict__ y, int nn) {
  int i = blockIdx.x * blockDim.x + threadIdx.x;
  if (i < nn) y[i] = f2bf(x[i]);
}

__global__ void k_seti(int* p, int v) { *p = v; }

// dense hash: flat voxel -> max(row index)
__global__ void k_table(const int* __restrict__ c, const int* __restrict__ v,
                        int* __restrict__ tbl, int gz, int gy, int gx, int n) {
  int r = blockIdx.x * blockDim.x + threadIdx.x;
  if (r >= n) return;
  if (v != nullptr && v[r] == 0) return;
  const int4 cc = ((const int4*)c)[r];
  if (cc.y < 0 || cc.y >= gz || cc.z < 0 || cc.z >= gy || cc.w < 0 || cc.w >= gx) return;
  atomicMax(&tbl[(cc.y * gy + cc.z) * gx + cc.w], r);
}

__global__ void k_down(const int* __restrict__ c, int sz, int sy, int sx,
                       int* __restrict__ q, int n) {
  int r = blockIdx.x * blockDim.x + threadIdx.x;
  if (r >= n) return;
  int4 cc = ((const int4*)c)[r];
  cc.y /= sz; cc.z /= sy; cc.w /= sx;
  ((int4*)q)[r] = cc;
}

__global__ void k_canon(const int* __restrict__ c, const int* __restrict__ v,
                        const int* __restrict__ tbl, int gz, int gy, int gx,
                        int* __restrict__ vout, int* __restrict__ cnt, int n) {
  int r = blockIdx.x * blockDim.x + threadIdx.x;
  if (r >= n) return;
  int ok = (v == nullptr) || (v[r] != 0);
  if (ok) {
    const int4 cc = ((const int4*)c)[r];
    ok = (cc.y >= 0 && cc.y < gz && cc.z >= 0 && cc.z < gy && cc.w >= 0 && cc.w < gx);
    if (ok) ok = (tbl[(cc.y * gy + cc.z) * gx + cc.w] == r);
  }
  vout[r] = ok ? 1 : 0;
  if (ok) atomicAdd(cnt, 1);
}

// weights (ntaps,cin,cout) f32 -> bf16 fragments in contiguous-K WMMA B order (K = 32*kt + 16*half + j):
// frag[((tap*KT+kt)*NT+nt)*512 + lane*16 + j], lane = (K-half<<4)|col, j -> K offset
__global__ void k_wfrag(const float* __restrict__ W, u16* __restrict__ frag,
                        int cin, int cw, int ntaps, int KT, int NT) {
  const int t = blockIdx.x * blockDim.x + threadIdx.x;
  const int total = ntaps * KT * NT * 32;
  if (t >= total) return;
  const int lane = t & 31;
  int f = t >> 5;
  const int nt = f % NT; f /= NT;
  const int kt = f % KT; f /= KT;
  const int tap = f;
  const int col = lane & 15, half = lane >> 4;
  u16* dst = frag + ((size_t)((tap * KT + kt) * NT + nt) * 512) + lane * 16;
  for (int j = 0; j < 16; ++j) {
    const int K = (kt << 5) + (half << 4) + j;
    const float w = (K < cin) ? W[((size_t)tap * cin + K) * cw + col] : 0.f;
    dst[j] = f2bf(w);
  }
}

// ---------------- WMMA sparse-conv kernel ----------------
// One wave (32 lanes) per 16-row M-tile. Gathers neighbor rows into LDS (bf16),
// then D = A(16x32 bf16) x B(32x16 bf16) + C via v_wmma_f32_16x16x32_bf16.
template<int KT, int NT>
__global__ __launch_bounds__(32)
void k_conv(const u16* __restrict__ xin, const int* __restrict__ oc,
            const int* __restrict__ vm, const int* __restrict__ tbl,
            int gz, int gy, int gx, int sz, int sy, int sx,
            const u16* __restrict__ wf, int cin, int ntaps, int use27,
            float* __restrict__ out, float* __restrict__ colsum,
            float* __restrict__ colsq, int n) {
  constexpr int SEGS = KT * 4;          // uint4 (8x bf16) segments per row
  __shared__ uint4 sA[16 * SEGS];       // 16 rows x KT*32 bf16
  __shared__ int   s_nb[16];

  const int lane = threadIdx.x;
  const int m    = lane & 15;
  const int half = lane >> 4;
  const int row0 = blockIdx.x * 16;
  const int segsv = cin >> 3;           // valid segments (cin may be < KT*32)

  v8f acc[NT];
#pragma unroll
  for (int t = 0; t < NT; ++t)
#pragma unroll
    for (int e = 0; e < 8; ++e) acc[t][e] = 0.0f;

  for (int tap = 0; tap < ntaps; ++tap) {
    const int dz = use27 ? c_off27[tap * 3 + 0] : c_offz[tap * 3 + 0];
    const int dy = use27 ? c_off27[tap * 3 + 1] : c_offz[tap * 3 + 1];
    const int dx = use27 ? c_off27[tap * 3 + 2] : c_offz[tap * 3 + 2];
    if (tap + 1 < ntaps)   // gfx1250 global_prefetch_b8 on next tap's weights
      __builtin_prefetch(wf + (size_t)(tap + 1) * KT * NT * 512, 0, 1);

    __syncthreads();
    if (lane < 16) {       // lanes 0..15 resolve one output row each
      const int row = row0 + lane;
      int idx = -1;
      if (row < n && (vm == nullptr || vm[row] != 0)) {
        const int4 cc = ((const int4*)oc)[row];
        const int z = cc.y * sz + dz, y = cc.z * sy + dy, x = cc.w * sx + dx;
        if (z >= 0 && z < gz && y >= 0 && y < gy && x >= 0 && x < gx)
          idx = tbl[(z * gy + y) * gx + x];
      }
      s_nb[lane] = idx;
    }
    __syncthreads();

    // cooperative gather of 16 neighbor rows (bf16) into LDS, zero when missing
    for (int i = lane; i < 16 * SEGS; i += 32) {
      const int row = i / SEGS, seg = i - row * SEGS;
      const int src = s_nb[row];
      uint4 val = make_uint4(0u, 0u, 0u, 0u);
      if (src >= 0 && seg < segsv)
        val = ((const uint4*)(xin + (size_t)src * cin))[seg];
      sA[row * SEGS + seg] = val;
    }
    __syncthreads();

    const uint4* wq = (const uint4*)(wf + (size_t)tap * KT * NT * 512);
#pragma unroll
    for (int kt = 0; kt < KT; ++kt) {
      // A fragment: lanes 0-15 -> K {kb..kb+15} of row m (contiguous, matches B)
      Frag16 a;
      const int kb = (kt << 5) + (half << 4);
      const uint4* arow = &sA[m * SEGS];
      a.q[0] = arow[kb >> 3];
      a.q[1] = arow[(kb + 8) >> 3];
#pragma unroll
      for (int nt = 0; nt < NT; ++nt) {
        Frag16 b;
        const uint4* bp = wq + (kt * NT + nt) * 64 + lane * 2;
        b.q[0] = bp[0];
        b.q[1] = bp[1];
        acc[nt] = __builtin_amdgcn_wmma_f32_16x16x32_bf16(
            false, a.v, false, b.v, (short)0, acc[nt], false, false);
      }
    }
  }

  // epilogue: masked store + fused per-column BN statistics
  const int cw = NT * 16;
#pragma unroll
  for (int nt = 0; nt < NT; ++nt) {
    const int col = (nt << 4) + m;
    float s1 = 0.f, s2 = 0.f;
#pragma unroll
    for (int r = 0; r < 8; ++r) {
      const int row = row0 + (half << 3) + r;   // C layout: M = r + 8*half
      if (row < n) {
        float v = acc[nt][r];
        if (vm != nullptr && vm[row] == 0) v = 0.f;
        out[(size_t)row * cw + col] = v;
        s1 += v; s2 += v * v;
      }
    }
    atomicAdd(&colsum[col], s1);
    atomicAdd(&colsq[col], s2);
  }
}

// BN (training stats over valid rows) + mask + optional residual + ReLU,
// also produces bf16 copy for the next gather stage
__global__ void k_bn_apply(const float* __restrict__ x, const float* __restrict__ res,
                           const int* __restrict__ vm,
                           const float* __restrict__ colsum, const float* __restrict__ colsq,
                           const int* __restrict__ cnt,
                           const float* __restrict__ gamma, const float* __restrict__ beta,
                           float* __restrict__ y, u16* __restrict__ ybf, int n, int c) {
  const int i = blockIdx.x * blockDim.x + threadIdx.x;
  if (i >= n * c) return;
  const int col = i % c;
  const int row = i / c;
  const int cv = *cnt;
  const float nv = (float)(cv < 1 ? 1 : cv);
  const float mu = colsum[col] / nv;
  float var = colsq[col] / nv - mu * mu;
  var = var < 0.f ? 0.f : var;
  float o = (x[i] - mu) * rsqrtf(var + 1e-5f) * gamma[col] + beta[col];
  if (vm != nullptr && vm[row] == 0) o = 0.f;
  if (res != nullptr) o += res[i];
  o = fmaxf(o, 0.f);
  y[i] = o;
  if (ybf != nullptr) ybf[i] = f2bf(o);
}

// scatter (B,2,188,188,128) -> (B,256,188,188): out[(col*gz+z)*gy*gx + y*gx + x]
__global__ void k_scatter(const float* __restrict__ x, const int* __restrict__ c,
                          const int* __restrict__ v, float* __restrict__ out,
                          int n, int ch, int gz, int gy, int gx) {
  const int i = blockIdx.x * blockDim.x + threadIdx.x;
  if (i >= n * ch) return;
  const int row = i / ch, col = i - row * ch;
  if (v[row] == 0) return;
  const int4 cc = ((const int4*)c)[row];
  const size_t o = (((size_t)col * gz + cc.y) * gy + cc.z) * gx + cc.w;
  atomicAdd(&out[o], x[i]);
}

// ---------------- host orchestration ----------------

struct LayerD { int w, g, b, cin, cout, ntaps; };
// d_in leaf indices assuming depth-first insertion-order flattening of params
static const LayerD LD[21] = {
  {2, 3, 4, 128, 16, 27},                                            // conv_input
  {5, 6, 7, 16, 16, 27},  {8, 9, 10, 16, 16, 27},                    // conv1 b0
  {11, 12, 13, 16, 16, 27}, {14, 15, 16, 16, 16, 27},                // conv1 b1
  {17, 18, 19, 16, 32, 27},                                          // conv2 W
  {20, 21, 22, 32, 32, 27}, {23, 24, 25, 32, 32, 27},                // conv2 b0
  {26, 27, 28, 32, 32, 27}, {29, 30, 31, 32, 32, 27},                // conv2 b1
  {32, 33, 34, 32, 64, 27},                                          // conv3 W
  {35, 36, 37, 64, 64, 27}, {38, 39, 40, 64, 64, 27},                // conv3 b0
  {41, 42, 43, 64, 64, 27}, {44, 45, 46, 64, 64, 27},                // conv3 b1
  {47, 48, 49, 64, 128, 27},                                         // conv4 W
  {50, 51, 52, 128, 128, 27}, {53, 54, 55, 128, 128, 27},            // conv4 b0
  {56, 57, 58, 128, 128, 27}, {59, 60, 61, 128, 128, 27},            // conv4 b1
  {62, 63, 64, 128, 128, 3},                                         // extra
};

static void launch_conv(hipStream_t st, int KT, int NT, int grid,
                        const u16* xin, const int* oc, const int* vm, const int* tbl,
                        const int* g, int sz, int sy, int sx,
                        const u16* wf, int cin, int ntaps,
                        float* out, float* colsum, float* colsq, int n) {
  const int use27 = (ntaps == 27) ? 1 : 0;
#define LC(kt, nt)                                                                   \
  if (KT == kt && NT == nt) {                                                        \
    k_conv<kt, nt><<<grid, 32, 0, st>>>(xin, oc, vm, tbl, g[0], g[1], g[2],          \
                                        sz, sy, sx, wf, cin, ntaps, use27,           \
                                        out, colsum, colsq, n);                      \
    return;                                                                          \
  }
  LC(4, 1) LC(1, 1) LC(1, 2) LC(1, 4) LC(2, 4) LC(2, 8) LC(4, 8)
#undef LC
}

static void conv_bn(hipStream_t st, const u16* xin, const int* oc, const int* vm,
                    const int* tbl, const int* g, int sz, int sy, int sx,
                    const u16* wf, int cin, int cw, int ntaps,
                    const float* gamma, const float* beta, const int* cnt,
                    const float* res, float* raw, float* y, u16* ybf,
                    float* colsum, float* colsq, int n) {
  hipMemsetAsync(colsum, 0, 512, st);
  hipMemsetAsync(colsq, 0, 512, st);
  const int cinp = cin < 32 ? 32 : cin;
  launch_conv(st, cinp / 32, cw / 16, (n + 15) / 16, xin, oc, vm, tbl, g, sz, sy, sx,
              wf, cin, ntaps, raw, colsum, colsq, n);
  const int tot = n * cw;
  k_bn_apply<<<(tot + 255) / 256, 256, 0, st>>>(raw, res, vm, colsum, colsq, cnt,
                                                gamma, beta, y, ybf, n, cw);
}

static void basic_block(hipStream_t st, void* const* d_in,
                        const LayerD& L1, const u16* f1, const LayerD& L2, const u16* f2,
                        const int* oc, const int* vm, const int* tbl, const int* g,
                        const int* cnt, float* x, u16* xb,
                        float* raw, float* t, u16* tb,
                        float* colsum, float* colsq, int n) {
  conv_bn(st, xb, oc, vm, tbl, g, 1, 1, 1, f1, L1.cin, L1.cout, 27,
          (const float*)d_in[L1.g], (const float*)d_in[L1.b], cnt, nullptr,
          raw, t, tb, colsum, colsq, n);
  conv_bn(st, tb, oc, vm, tbl, g, 1, 1, 1, f2, L2.cin, L2.cout, 27,
          (const float*)d_in[L2.g], (const float*)d_in[L2.b], cnt, x,
          raw, x, xb, colsum, colsq, n);
}

extern "C" void kernel_launch(void* const* d_in, const int* in_sizes, int n_in,
                              void* d_out, int out_size, void* d_ws, size_t ws_size,
                              hipStream_t stream) {
  (void)in_sizes; (void)n_in; (void)ws_size;
  const int N = 40000;
  const float* feats = (const float*)d_in[0];
  const int*   c1    = (const int*)d_in[1];

  static const int G1[3] = {40, 1504, 1504}, G2[3] = {20, 752, 752},
                   G3[3] = {10, 376, 376},   G4[3] = {5, 188, 188},
                   G5[3] = {2, 188, 188};
  const size_t T1 = (size_t)40 * 1504 * 1504, T2 = (size_t)20 * 752 * 752,
               T3 = (size_t)10 * 376 * 376,  T4 = (size_t)5 * 188 * 188,
               T5 = (size_t)2 * 188 * 188;

  char* wp = (char*)d_ws;
  auto carve = [&](size_t bytes) -> void* {
    void* p = (void*)wp;
    wp += (bytes + 255) & ~(size_t)255;
    return p;
  };

  int* tblA = (int*)carve(T1 * 4);
  int* tblB = (int*)carve(T2 * 4);
  int* tblC = (int*)carve(T3 * 4);
  int* tblD = (int*)carve(T4 * 4);
  int* tblE = (int*)carve(T5 * 4);
  int* cB = (int*)carve((size_t)N * 16);
  int* cC = (int*)carve((size_t)N * 16);
  int* cD = (int*)carve((size_t)N * 16);
  int* cE = (int*)carve((size_t)N * 16);
  int* vB = (int*)carve((size_t)N * 4);
  int* vC = (int*)carve((size_t)N * 4);
  int* vD = (int*)carve((size_t)N * 4);
  int* vE = (int*)carve((size_t)N * 4);
  int* cnts = (int*)carve(8 * 4);
  float* colsum = (float*)carve(512);
  float* colsq  = (float*)carve(512);
  float* fx0 = (float*)carve((size_t)N * 128 * 4);
  float* fx1 = (float*)carve((size_t)N * 128 * 4);
  float* fx2 = (float*)carve((size_t)N * 128 * 4);
  u16* fb0 = (u16*)carve((size_t)N * 128 * 2);
  u16* fb1 = (u16*)carve((size_t)N * 128 * 2);

  // pre-swizzle all conv weights into bf16 WMMA fragment layout
  u16* frag[21];
  for (int i = 0; i < 21; ++i) {
    const LayerD& L = LD[i];
    const int cinp = L.cin < 32 ? 32 : L.cin;
    const int KT = cinp / 32, NT = L.cout / 16;
    frag[i] = (u16*)carve((size_t)L.ntaps * KT * NT * 512 * 2);
    const int tot = L.ntaps * KT * NT * 32;
    k_wfrag<<<(tot + 63) / 64, 64, 0, stream>>>((const float*)d_in[L.w], frag[i],
                                                L.cin, L.cout, L.ntaps, KT, NT);
  }

  k_f2bf<<<(N * 128 + 255) / 256, 256, 0, stream>>>(feats, fb0, N * 128);
  k_seti<<<1, 1, 0, stream>>>(cnts, N);

  hipMemsetAsync(tblA, 0xFF, T1 * 4, stream);
  k_table<<<(N + 255) / 256, 256, 0, stream>>>(c1, nullptr, tblA, G1[0], G1[1], G1[2], N);

  // conv_input (128->16) + conv1 blocks at level 1
  conv_bn(stream, fb0, c1, nullptr, tblA, G1, 1, 1, 1, frag[0], 128, 16, 27,
          (const float*)d_in[LD[0].g], (const float*)d_in[LD[0].b], cnts, nullptr,
          fx1, fx0, fb0, colsum, colsq, N);
  basic_block(stream, d_in, LD[1], frag[1], LD[2], frag[2], c1, nullptr, tblA, G1, cnts,
              fx0, fb0, fx1, fx2, fb1, colsum, colsq, N);
  basic_block(stream, d_in, LD[3], frag[3], LD[4], frag[4], c1, nullptr, tblA, G1, cnts,
              fx0, fb0, fx1, fx2, fb1, colsum, colsq, N);

  // ---- stage conv2 (stride 2): level1 -> level2 ----
  k_down<<<(N + 255) / 256, 256, 0, stream>>>(c1, 2, 2, 2, cB, N);
  hipMemsetAsync(tblB, 0xFF, T2 * 4, stream);
  k_table<<<(N + 255) / 256, 256, 0, stream>>>(cB, nullptr, tblB, G2[0], G2[1], G2[2], N);
  hipMemsetAsync(cnts + 1, 0, 4, stream);
  k_canon<<<(N + 255) / 256, 256, 0, stream>>>(cB, nullptr, tblB, G2[0], G2[1], G2[2], vB, cnts + 1, N);
  conv_bn(stream, fb0, cB, vB, tblA, G1, 2, 2, 2, frag[5], 16, 32, 27,
          (const float*)d_in[LD[5].g], (const float*)d_in[LD[5].b], cnts + 1, nullptr,
          fx1, fx0, fb0, colsum, colsq, N);
  basic_block(stream, d_in, LD[6], frag[6], LD[7], frag[7], cB, vB, tblB, G2, cnts + 1,
              fx0, fb0, fx1, fx2, fb1, colsum, colsq, N);
  basic_block(stream, d_in, LD[8], frag[8], LD[9], frag[9], cB, vB, tblB, G2, cnts + 1,
              fx0, fb0, fx1, fx2, fb1, colsum, colsq, N);

  // ---- stage conv3 (stride 2): level2 -> level3 ----
  k_down<<<(N + 255) / 256, 256, 0, stream>>>(cB, 2, 2, 2, cC, N);
  hipMemsetAsync(tblC, 0xFF, T3 * 4, stream);
  k_table<<<(N + 255) / 256, 256, 0, stream>>>(cC, vB, tblC, G3[0], G3[1], G3[2], N);
  hipMemsetAsync(cnts + 2, 0, 4, stream);
  k_canon<<<(N + 255) / 256, 256, 0, stream>>>(cC, vB, tblC, G3[0], G3[1], G3[2], vC, cnts + 2, N);
  conv_bn(stream, fb0, cC, vC, tblB, G2, 2, 2, 2, frag[10], 32, 64, 27,
          (const float*)d_in[LD[10].g], (const float*)d_in[LD[10].b], cnts + 2, nullptr,
          fx1, fx0, fb0, colsum, colsq, N);
  basic_block(stream, d_in, LD[11], frag[11], LD[12], frag[12], cC, vC, tblC, G3, cnts + 2,
              fx0, fb0, fx1, fx2, fb1, colsum, colsq, N);
  basic_block(stream, d_in, LD[13], frag[13], LD[14], frag[14], cC, vC, tblC, G3, cnts + 2,
              fx0, fb0, fx1, fx2, fb1, colsum, colsq, N);

  // ---- stage conv4 (stride 2): level3 -> level4 ----
  k_down<<<(N + 255) / 256, 256, 0, stream>>>(cC, 2, 2, 2, cD, N);
  hipMemsetAsync(tblD, 0xFF, T4 * 4, stream);
  k_table<<<(N + 255) / 256, 256, 0, stream>>>(cD, vC, tblD, G4[0], G4[1], G4[2], N);
  hipMemsetAsync(cnts + 3, 0, 4, stream);
  k_canon<<<(N + 255) / 256, 256, 0, stream>>>(cD, vC, tblD, G4[0], G4[1], G4[2], vD, cnts + 3, N);
  conv_bn(stream, fb0, cD, vD, tblC, G3, 2, 2, 2, frag[15], 64, 128, 27,
          (const float*)d_in[LD[15].g], (const float*)d_in[LD[15].b], cnts + 3, nullptr,
          fx1, fx0, fb0, colsum, colsq, N);
  basic_block(stream, d_in, LD[16], frag[16], LD[17], frag[17], cD, vD, tblD, G4, cnts + 3,
              fx0, fb0, fx1, fx2, fb1, colsum, colsq, N);
  basic_block(stream, d_in, LD[18], frag[18], LD[19], frag[19], cD, vD, tblD, G4, cnts + 3,
              fx0, fb0, fx1, fx2, fb1, colsum, colsq, N);

  // ---- extra conv (stride (3,1,1), 3 taps): level4 -> level5 ----
  k_down<<<(N + 255) / 256, 256, 0, stream>>>(cD, 3, 1, 1, cE, N);
  hipMemsetAsync(tblE, 0xFF, T5 * 4, stream);
  k_table<<<(N + 255) / 256, 256, 0, stream>>>(cE, vD, tblE, G5[0], G5[1], G5[2], N);
  hipMemsetAsync(cnts + 4, 0, 4, stream);
  k_canon<<<(N + 255) / 256, 256, 0, stream>>>(cE, vD, tblE, G5[0], G5[1], G5[2], vE, cnts + 4, N);
  conv_bn(stream, fb0, cE, vE, tblD, G4, 3, 1, 1, frag[20], 128, 128, 3,
          (const float*)d_in[LD[20].g], (const float*)d_in[LD[20].b], cnts + 4, nullptr,
          fx1, fx0, nullptr, colsum, colsq, N);

  // ---- dense scatter to (B,256,188,188) ----
  hipMemsetAsync(d_out, 0, (size_t)out_size * sizeof(float), stream);
  k_scatter<<<(N * 128 + 255) / 256, 256, 0, stream>>>(fx0, cE, vE, (float*)d_out,
                                                       N, 128, G5[0], G5[1], G5[2]);
}
